// QEffDeepseekV3Attention_23871428231437
// MI455X (gfx1250) — matrix operand order repaired and checked by reading, not compiled, not run on hardware
//
#include <hip/hip_runtime.h>
#include <hip/hip_bf16.h>
#include <math.h>

typedef __attribute__((ext_vector_type(16))) __bf16 v16bf;
typedef __attribute__((ext_vector_type(8)))  __bf16 v8bf;
typedef __attribute__((ext_vector_type(8)))  float  v8f;

#define B_      2
#define S_      2048
#define HID_    2048
#define NH_     16
#define QD_     192
#define DN_     128
#define DR_     64
#define DV_     128
#define QLORA_  1536
#define KVLORA_ 512
#define CKVW_   (KVLORA_ + DR_)          // 576
#define KVW_    (NH_ * (DN_ + DV_))      // 4096
#define QW_     (NH_ * QD_)              // 3072
#define SCALE_  0.07216878364870323f     // 1/sqrt(192)

// ---------------------------------------------------------------------------
// Fragment loaders for CDNA5 WMMA 16x16x32 bf16; this code's contiguous k order (A and B alike)
// A (16x32, row-major src, ld = K):
//   lane: M = lane&15 ; elems 0..15 -> K = kaoff+0..15
//   kaoff = (lane>=16) ? 16 : 0
// B (32x16, src stored transposed: Bt[N][K], ld = K):
//   lane: N = lane&15 ; elems 0..15 -> K = kboff+0..15 ; kboff = (lane>=16) ? 16 : 0
// C/D f32 16x16: VGPR r -> M = r + ((lane>=16)?8:0), N = lane&15
// ---------------------------------------------------------------------------
__device__ __forceinline__ v16bf pack16(v8bf lo, v8bf hi) {
  return __builtin_shufflevector(lo, hi, 0,1,2,3,4,5,6,7,8,9,10,11,12,13,14,15);
}

__device__ __forceinline__ v16bf load_a_frag(const __bf16* __restrict__ A,
                                             int lda, int m0, int k0) {
  const int lane = threadIdx.x & 31;
  const __bf16* p = A + (size_t)(m0 + (lane & 15)) * lda + k0 + ((lane >> 4) << 4);
  v8bf lo = *(const v8bf*)p;
  v8bf hi = *(const v8bf*)(p + 8);
  return pack16(lo, hi);
}

__device__ __forceinline__ v16bf load_b_frag(const __bf16* __restrict__ Bt,
                                             int ldb, int n0, int k0) {
  const int lane = threadIdx.x & 31;
  return *(const v16bf*)(Bt + (size_t)(n0 + (lane & 15)) * ldb + k0 + ((lane >> 4) << 4));
}

#define WMMA_BF16(a, b, c) \
  __builtin_amdgcn_wmma_f32_16x16x32_bf16(false, (a), false, (b), (short)0, (c), false, false)

// ---------------------------------------------------------------------------
// Generic bf16 WMMA GEMM: C[M,N] (f32) = A[M,K] (bf16, row-major) x Bt[N,K]^T
// 8 waves / block; wave tile 32(M) x 64(N); block tile 256 x 64.
// 8 WMMAs per k-step per wave; B fragments reused across 2 A fragments.
// M % 256 == 0, N % 64 == 0, K % 32 == 0 (all shapes here satisfy this).
// ---------------------------------------------------------------------------
__global__ __launch_bounds__(256) void gemm_bf16_wmma(
    const __bf16* __restrict__ A, const __bf16* __restrict__ Bt,
    float* __restrict__ C, int M, int N, int K) {
  const int lane = threadIdx.x & 31;
  const int wave = threadIdx.x >> 5;
  const int m0 = blockIdx.y * 256 + wave * 32;
  const int n0 = blockIdx.x * 64;

  v8f acc[2][4];
#pragma unroll
  for (int s = 0; s < 2; ++s)
#pragma unroll
    for (int j = 0; j < 4; ++j) acc[s][j] = v8f{};

  const __bf16* pfA = A + (size_t)(m0 + (lane & 15)) * K;
  const __bf16* pfB = Bt + (size_t)(n0 + (lane & 15)) * K;

  for (int k0 = 0; k0 < K; k0 += 32) {
    // speculative prefetch of next k-tile (global_prefetch_b8; dropped if OOB)
    __builtin_prefetch(pfA + k0 + 32, 0, 0);
    __builtin_prefetch(pfB + k0 + 32, 0, 0);

    v16bf a0 = load_a_frag(A, K, m0,      k0);
    v16bf a1 = load_a_frag(A, K, m0 + 16, k0);
#pragma unroll
    for (int j = 0; j < 4; ++j) {
      v16bf b = load_b_frag(Bt, K, n0 + j * 16, k0);
      acc[0][j] = WMMA_BF16(a0, b, acc[0][j]);
      acc[1][j] = WMMA_BF16(a1, b, acc[1][j]);
    }
  }

#pragma unroll
  for (int s = 0; s < 2; ++s) {
    const int rowC = m0 + s * 16 + ((lane >> 4) << 3);
#pragma unroll
    for (int j = 0; j < 4; ++j) {
      const int col = n0 + j * 16 + (lane & 15);
#pragma unroll
      for (int r = 0; r < 8; ++r)
        C[(size_t)(rowC + r) * N + col] = acc[s][j][r];
    }
  }
}

// ---------------------------------------------------------------------------
// Elementwise helpers
// ---------------------------------------------------------------------------
__global__ void convert_f32_bf16(const float* __restrict__ in,
                                 __bf16* __restrict__ out, size_t n) {
  size_t i = (size_t)blockIdx.x * blockDim.x + threadIdx.x;
  if (i < n) out[i] = (__bf16)in[i];
}

// out[c*R + r] = in[r*C + c]  (fp32 -> bf16 transpose for weight matrices)
__global__ void transpose_f32_bf16(const float* __restrict__ in,
                                   __bf16* __restrict__ out, int R, int C) {
  size_t i = (size_t)blockIdx.x * blockDim.x + threadIdx.x;
  if (i >= (size_t)R * C) return;
  int r = (int)(i / C), c = (int)(i % C);
  out[(size_t)c * R + r] = (__bf16)in[i];
}

// RMSNorm: one block per row; in row stride = in_stride, out row stride = width
__global__ __launch_bounds__(256) void rmsnorm_bf16(
    const float* __restrict__ in, const float* __restrict__ w,
    __bf16* __restrict__ out, int width, int in_stride) {
  __shared__ float red[256];
  const int row = blockIdx.x;
  const float* p = in + (size_t)row * in_stride;
  float s = 0.f;
  for (int i = threadIdx.x; i < width; i += 256) { float v = p[i]; s += v * v; }
  red[threadIdx.x] = s;
  __syncthreads();
  for (int off = 128; off > 0; off >>= 1) {
    if ((int)threadIdx.x < off) red[threadIdx.x] += red[threadIdx.x + off];
    __syncthreads();
  }
  const float rr = rsqrtf(red[0] / (float)width + 1e-6f);
  for (int i = threadIdx.x; i < width; i += 256)
    out[(size_t)row * width + i] = (__bf16)(p[i] * rr * w[i]);
}

// ---------------------------------------------------------------------------
// RoPE / layout prep
// ---------------------------------------------------------------------------
__device__ __forceinline__ float rope_val(const float* pe, int i, float pos) {
  const int j = (i < 32) ? i : i - 32;
  const float freq = powf(10000.0f, -(2.0f * (float)j) / 64.0f);
  const float ang = pos * freq;
  const float c = cosf(ang), sn = sinf(ang);
  const float e = pe[2 * j], o = pe[2 * j + 1];
  return (i < 32) ? (e * c - o * sn) : (o * c + e * sn);
}

__global__ void prep_q_rope(const float* __restrict__ q,
                            const int* __restrict__ pos_ids,
                            __bf16* __restrict__ Qb) {
  size_t idx = (size_t)blockIdx.x * blockDim.x + threadIdx.x;
  const size_t total = (size_t)B_ * NH_ * S_ * QD_;
  if (idx >= total) return;
  const int d = (int)(idx % QD_);
  size_t t = idx / QD_;
  const int s = (int)(t % S_); t /= S_;
  const int h = (int)(t % NH_);
  const int b = (int)(t / NH_);
  const float* qrow = q + (size_t)(b * S_ + s) * QW_ + h * QD_;
  float val;
  if (d < DN_) {
    val = qrow[d];
  } else {
    const float pos = (float)pos_ids[(size_t)b * S_ + s];
    val = rope_val(qrow + DN_, d - DN_, pos);
  }
  Qb[idx] = (__bf16)val;
}

__global__ void prep_k_rope(const float* __restrict__ kv,
                            const float* __restrict__ ckv,
                            const int* __restrict__ pos_ids,
                            __bf16* __restrict__ Kb) {
  size_t idx = (size_t)blockIdx.x * blockDim.x + threadIdx.x;
  const size_t total = (size_t)B_ * NH_ * S_ * QD_;
  if (idx >= total) return;
  const int d = (int)(idx % QD_);
  size_t t = idx / QD_;
  const int s = (int)(t % S_); t /= S_;
  const int h = (int)(t % NH_);
  const int b = (int)(t / NH_);
  float val;
  if (d < DN_) {
    val = kv[(size_t)(b * S_ + s) * KVW_ + h * (DN_ + DV_) + d];
  } else {
    const float pos = (float)pos_ids[(size_t)b * S_ + s];
    val = rope_val(ckv + (size_t)(b * S_ + s) * CKVW_ + KVLORA_, d - DN_, pos);
  }
  Kb[idx] = (__bf16)val;
}

// kv fp32 -> Vt bf16 [B,NH,DV,S]  (transposed so PV B-fragments are contiguous)
__global__ void prep_v(const float* __restrict__ kv, __bf16* __restrict__ Vt) {
  size_t idx = (size_t)blockIdx.x * blockDim.x + threadIdx.x;
  const size_t total = (size_t)B_ * NH_ * DV_ * S_;
  if (idx >= total) return;
  const int s = (int)(idx % S_);
  size_t t = idx / S_;
  const int d = (int)(t % DV_); t /= DV_;
  const int h = (int)(t % NH_);
  const int b = (int)(t / NH_);
  Vt[idx] = (__bf16)kv[(size_t)(b * S_ + s) * KVW_ + h * (DN_ + DV_) + DN_ + d];
}

// ---------------------------------------------------------------------------
// Flash attention, causal. 4 waves/block; wave = one 16-row q tile; k tiles of 32.
// grid = (S/64, B*NH). Output attn bf16 [B*S, NH*DV].
// ---------------------------------------------------------------------------
__global__ __launch_bounds__(128) void mla_attention_wmma(
    const __bf16* __restrict__ Qb, const __bf16* __restrict__ Kb,
    const __bf16* __restrict__ Vt, __bf16* __restrict__ attnOut) {
  __shared__ __bf16 Plds[4][16][48];   // per-wave P tile (16x32, padded row stride)

  const int lane = threadIdx.x & 31;
  const int wave = threadIdx.x >> 5;
  const int bh = blockIdx.y;
  const int b = bh >> 4, h = bh & 15;
  const int q0 = blockIdx.x * 64 + wave * 16;
  const int hi8 = (lane >> 4) << 3;      // C-fragment row offset for upper lanes
  const int kaoff = (lane >> 4) << 4;    // A-fragment K offset
  const int kboff = (lane >> 4) << 4;    // B-fragment K offset

  const __bf16* Qp = Qb + (size_t)bh * S_ * QD_;
  const __bf16* Kp = Kb + (size_t)bh * S_ * QD_;
  const __bf16* Vp = Vt + (size_t)bh * DV_ * S_;

  // Preload the 6 Q A-fragments (d = 0..191 in chunks of 32)
  v16bf qf[6];
#pragma unroll
  for (int d = 0; d < 6; ++d) qf[d] = load_a_frag(Qp, QD_, q0, d * 32);

  v8f o[8];
  float m[8], l[8];
#pragma unroll
  for (int r = 0; r < 8; ++r) { o[r] = v8f{}; m[r] = -1e30f; l[r] = 0.f; }

  const int numKt = (q0 + 47) >> 5;      // cover columns up to q0+15 (causal)

  for (int kt = 0; kt < numKt; ++kt) {
    const int kb0 = kt * 32;

    // prefetch next K tile rows for this lane (speculative)
    __builtin_prefetch(Kp + (size_t)(kb0 + 32 + (lane & 15)) * QD_, 0, 0);

    // ---- S = Q K^T for 16x32 tile: two C fragments -----------------------
    v8f s0{}, s1{};
#pragma unroll
    for (int d = 0; d < 6; ++d) {
      v16bf kbA = load_b_frag(Kp, QD_, kb0,      d * 32);
      s0 = WMMA_BF16(qf[d], kbA, s0);
      v16bf kbB = load_b_frag(Kp, QD_, kb0 + 16, d * 32);
      s1 = WMMA_BF16(qf[d], kbB, s1);
    }

    // ---- scale, causal mask, online softmax ------------------------------
    float pf[8];
    const int c0 = kb0 + (lane & 15);
#pragma unroll
    for (int r = 0; r < 8; ++r) {
      const int row = q0 + r + hi8;
      float x0 = (c0 > row)      ? -10000.f : s0[r] * SCALE_;
      float x1 = (c0 + 16 > row) ? -10000.f : s1[r] * SCALE_;
      float mx = fmaxf(x0, x1);
      mx = fmaxf(mx, __shfl_xor(mx, 1));
      mx = fmaxf(mx, __shfl_xor(mx, 2));
      mx = fmaxf(mx, __shfl_xor(mx, 4));
      mx = fmaxf(mx, __shfl_xor(mx, 8));
      const float mn = fmaxf(m[r], mx);
      const float fac = __expf(m[r] - mn);
      m[r] = mn;
      const float p0 = __expf(x0 - mn);
      const float p1 = __expf(x1 - mn);
      float rs = p0 + p1;
      rs += __shfl_xor(rs, 1);
      rs += __shfl_xor(rs, 2);
      rs += __shfl_xor(rs, 4);
      rs += __shfl_xor(rs, 8);
      l[r] = l[r] * fac + rs;
      pf[r] = fac;
      // stage P into LDS (bf16) for C->A fragment re-layout
      Plds[wave][r + hi8][lane & 15]        = (__bf16)p0;
      Plds[wave][r + hi8][16 + (lane & 15)] = (__bf16)p1;
    }

    // ---- read P back as an A fragment (16x32) ----------------------------
    const __bf16* pr = &Plds[wave][lane & 15][kaoff];
    v16bf pa = pack16(*(const v8bf*)pr, *(const v8bf*)(pr + 8));

    // ---- rescale accumulators, then O += P V -----------------------------
#pragma unroll
    for (int nt = 0; nt < 8; ++nt)
#pragma unroll
      for (int r = 0; r < 8; ++r) o[nt][r] *= pf[r];

#pragma unroll
    for (int nt = 0; nt < 8; ++nt) {
      v16bf vb = load_b_frag(Vp, S_, nt * 16, kb0);  // Vt[N=d][K=s]
      o[nt] = WMMA_BF16(pa, vb, o[nt]);
    }
  }

  // ---- normalize and write [B*S, NH*DV] bf16 -----------------------------
  float inv[8];
#pragma unroll
  for (int r = 0; r < 8; ++r) inv[r] = 1.0f / l[r];
#pragma unroll
  for (int nt = 0; nt < 8; ++nt)
#pragma unroll
    for (int r = 0; r < 8; ++r) {
      const int row = q0 + r + hi8;
      attnOut[(size_t)(b * S_ + row) * (NH_ * DV_) + h * DV_ + nt * 16 + (lane & 15)] =
          (__bf16)(o[nt][r] * inv[r]);
    }
}

// ---------------------------------------------------------------------------
// Host launcher
// ---------------------------------------------------------------------------
static inline unsigned nblk(size_t n) { return (unsigned)((n + 255) / 256); }

extern "C" void kernel_launch(void* const* d_in, const int* in_sizes, int n_in,
                              void* d_out, int out_size, void* d_ws, size_t ws_size,
                              hipStream_t stream) {
  (void)in_sizes; (void)n_in; (void)out_size; (void)ws_size;
  const float* hidden  = (const float*)d_in[0];
  /* d_in[1] attention_mask: causal, computed analytically */
  const int*   pos     = (const int*)d_in[2];
  const float* Wqa     = (const float*)d_in[3];
  const float* qa_lnw  = (const float*)d_in[4];
  const float* Wqb     = (const float*)d_in[5];
  const float* Wkva    = (const float*)d_in[6];
  const float* kva_lnw = (const float*)d_in[7];
  const float* Wkvb    = (const float*)d_in[8];
  const float* Wo      = (const float*)d_in[9];

  char* ws = (char*)d_ws;
  auto alloc = [&](size_t bytes) -> char* {
    char* p = ws;
    ws += (bytes + 255) & ~(size_t)255;
    return p;
  };

  const size_t nX   = (size_t)B_ * S_ * HID_;        // 8.39M
  const size_t nQA  = (size_t)B_ * S_ * QLORA_;      // 6.29M
  const size_t nQF  = (size_t)B_ * S_ * QW_;         // 12.6M
  const size_t nCKV = (size_t)B_ * S_ * CKVW_;       // 2.36M
  const size_t nCKN = (size_t)B_ * S_ * KVLORA_;
  const size_t nKV  = (size_t)B_ * S_ * KVW_;        // 16.8M
  const size_t nQK  = (size_t)B_ * NH_ * S_ * QD_;   // 12.6M
  const size_t nV   = (size_t)B_ * NH_ * DV_ * S_;   // 8.39M

  __bf16* Xb    = (__bf16*)alloc(nX * 2);
  __bf16* WqaT  = (__bf16*)alloc((size_t)HID_ * QLORA_ * 2);
  __bf16* WqbT  = (__bf16*)alloc((size_t)QLORA_ * QW_ * 2);
  __bf16* WkvaT = (__bf16*)alloc((size_t)HID_ * CKVW_ * 2);
  __bf16* WkvbT = (__bf16*)alloc((size_t)KVLORA_ * KVW_ * 2);
  __bf16* WoT   = (__bf16*)alloc((size_t)(NH_ * DV_) * HID_ * 2);
  float*  qa    = (float*)alloc(nQA * 4);
  __bf16* qan   = (__bf16*)alloc(nQA * 2);
  float*  qf    = (float*)alloc(nQF * 4);
  float*  ckv   = (float*)alloc(nCKV * 4);
  __bf16* ckvn  = (__bf16*)alloc(nCKN * 2);
  float*  kvf   = (float*)alloc(nKV * 4);
  __bf16* Qbm   = (__bf16*)alloc(nQK * 2);
  __bf16* Kbm   = (__bf16*)alloc(nQK * 2);
  __bf16* Vtm   = (__bf16*)alloc(nV * 2);
  __bf16* attn  = (__bf16*)alloc(nX * 2);

  // --- convert / transpose to bf16 ---
  convert_f32_bf16<<<nblk(nX), 256, 0, stream>>>(hidden, Xb, nX);
  transpose_f32_bf16<<<nblk((size_t)HID_ * QLORA_), 256, 0, stream>>>(Wqa, WqaT, HID_, QLORA_);
  transpose_f32_bf16<<<nblk((size_t)QLORA_ * QW_), 256, 0, stream>>>(Wqb, WqbT, QLORA_, QW_);
  transpose_f32_bf16<<<nblk((size_t)HID_ * CKVW_), 256, 0, stream>>>(Wkva, WkvaT, HID_, CKVW_);
  transpose_f32_bf16<<<nblk((size_t)KVLORA_ * KVW_), 256, 0, stream>>>(Wkvb, WkvbT, KVLORA_, KVW_);
  transpose_f32_bf16<<<nblk((size_t)(NH_ * DV_) * HID_), 256, 0, stream>>>(Wo, WoT, NH_ * DV_, HID_);

  const int Mrows = B_ * S_;  // 4096

  // --- q path: qa = X Wqa ; rms ; q = qa_n Wqb ---
  gemm_bf16_wmma<<<dim3(QLORA_ / 64, Mrows / 256), 256, 0, stream>>>(Xb, WqaT, qa, Mrows, QLORA_, HID_);
  rmsnorm_bf16<<<Mrows, 256, 0, stream>>>(qa, qa_lnw, qan, QLORA_, QLORA_);
  gemm_bf16_wmma<<<dim3(QW_ / 64, Mrows / 256), 256, 0, stream>>>(qan, WqbT, qf, Mrows, QW_, QLORA_);

  // --- kv path: ckv = X Wkva ; rms(c_kv) ; kv = ckv_n Wkvb ---
  gemm_bf16_wmma<<<dim3(CKVW_ / 64, Mrows / 256), 256, 0, stream>>>(Xb, WkvaT, ckv, Mrows, CKVW_, HID_);
  rmsnorm_bf16<<<Mrows, 256, 0, stream>>>(ckv, kva_lnw, ckvn, KVLORA_, CKVW_);
  gemm_bf16_wmma<<<dim3(KVW_ / 64, Mrows / 256), 256, 0, stream>>>(ckvn, WkvbT, kvf, Mrows, KVW_, KVLORA_);

  // --- RoPE + layout prep ---
  prep_q_rope<<<nblk(nQK), 256, 0, stream>>>(qf, pos, Qbm);
  prep_k_rope<<<nblk(nQK), 256, 0, stream>>>(kvf, ckv, pos, Kbm);
  prep_v<<<nblk(nV), 256, 0, stream>>>(kvf, Vtm);

  // --- flash attention ---
  mla_attention_wmma<<<dim3(S_ / 64, B_ * NH_), 128, 0, stream>>>(Qbm, Kbm, Vtm, attn);

  // --- output projection straight into d_out (fp32) ---
  gemm_bf16_wmma<<<dim3(HID_ / 64, Mrows / 256), 256, 0, stream>>>(attn, WoT, (float*)d_out,
                                                                  Mrows, HID_, NH_ * DV_);
}
